// Evaluate_6158983102660
// MI455X (gfx1250) — matrix operand order repaired and checked by reading, not compiled, not run on hardware
//
#include <hip/hip_runtime.h>
#include <hip/hip_bf16.h>
#include <math.h>

// Problem constants (from reference setup_inputs)
#define BB   2
#define CC   128
#define HH   96
#define WW   96
#define HWH  (HH * WW)        // 9216
#define NUM  48
#define KOUT 16               // num / FILTER_SIZE
#define TP   16               // pixels per pack/cost tile

typedef __attribute__((ext_vector_type(16))) _Float16 v16h;
typedef __attribute__((ext_vector_type(8)))  _Float16 v8h;
typedef __attribute__((ext_vector_type(8)))  float    v8f;

__device__ __forceinline__ int clampi(int v, int lo, int hi) {
    return v < lo ? lo : (v > hi ? hi : v);
}

// ---------------------------------------------------------------------------
// Kernel 0: pack/transpose both feature tensors to (B, HW, C) f16 so the
// gather kernel can use contiguous 16B vector loads. LDS-tiled so both the
// strided reads (along p) and the row writes (along c) are coalesced.
//   grid = (HW/TP, B), block = 256.
// ---------------------------------------------------------------------------
__global__ __launch_bounds__(256) void k_pack(
    const float* __restrict__ lf,   // (B, C, HW)
    const float* __restrict__ rf,   // (C, B*HW)
    _Float16* __restrict__ lf16,    // (B, HW, C)
    _Float16* __restrict__ rf16)    // (B, HW, C)
{
    const int tile = blockIdx.x;
    const int b    = blockIdx.y;
    const int p0   = tile * TP;
    const int tid  = threadIdx.x;

    __shared__ _Float16 t[TP][CC + 8];

    // ---- left features
#pragma unroll
    for (int e = 0; e < 8; ++e) {
        const int idx = tid + 256 * e;          // TP*CC = 2048 elements
        const int ch  = idx >> 4;
        const int i   = idx & 15;
        t[i][ch] = (_Float16)lf[(size_t)b * CC * HWH + (size_t)ch * HWH + p0 + i];
    }
    __syncthreads();
#pragma unroll
    for (int e = 0; e < 8; ++e) {
        const int idx = tid + 256 * e;
        const int i   = idx >> 7;
        const int ch  = idx & 127;
        lf16[((size_t)b * HWH + p0 + i) * CC + ch] = t[i][ch];
    }
    __syncthreads();

    // ---- right features
#pragma unroll
    for (int e = 0; e < 8; ++e) {
        const int idx = tid + 256 * e;
        const int ch  = idx >> 4;
        const int i   = idx & 15;
        t[i][ch] = (_Float16)rf[(size_t)ch * (BB * HWH) + (size_t)b * HWH + p0 + i];
    }
    __syncthreads();
#pragma unroll
    for (int e = 0; e < 8; ++e) {
        const int idx = tid + 256 * e;
        const int i   = idx >> 7;
        const int ch  = idx & 127;
        rf16[((size_t)b * HWH + p0 + i) * CC + ch] = t[i][ch];
    }
}

// ---------------------------------------------------------------------------
// Kernel 1: per 16-pixel tile, compute gather indices and match_cost via WMMA
// (diagonal trick: A rows = left features of 16 pixels, B cols = gathered
// right features of the same 16 pixels; diag(AxB) = cost[p, k]).
// Software-pipelined: the gather for slot k+1 is issued before the WMMA chain
// for slot k, so the scattered-L2 latency is hidden behind 4 WMMAs + stores.
//   grid = (HW/16, B), block = 32 (one wave; EXEC all-ones at every WMMA).
// ---------------------------------------------------------------------------
__global__ __launch_bounds__(32) void k_cost(
    const _Float16* __restrict__ lf16, // (B, HW, C)
    const _Float16* __restrict__ rf16, // (B, HW, C)
    const float* __restrict__ offx,    // (B, NUM, H, W)
    const float* __restrict__ offy,    // (B, NUM, H, W)
    float* __restrict__ cost,          // (B, NUM, HW)
    int*   __restrict__ inds)          // (B, HW, NUM)
{
    const int tile = blockIdx.x;
    const int b    = blockIdx.y;
    const int lane = threadIdx.x;
    const int col  = lane & 15;        // pixel within tile for this lane
    const bool hi  = lane >= 16;

    const int p  = tile * 16 + col;
    const int px = p % WW;
    const int py = p / WW;

    // ---- A fragments: 16x32 f16 per chunk, 4 chunks cover C=128.
    // ISA layout (16-bit A 16x32): lane = M (mod 16);
    //   lanes 0-15  hold K = {0..7, 16..23}, lanes 16-31 K = {8..15, 24..31}.
    const _Float16* lrow = lf16 + ((size_t)b * HWH + p) * CC;
    v16h A[4];
#pragma unroll
    for (int kk = 0; kk < 4; ++kk) {
        const int o = 32 * kk + (hi ? 8 : 0);
        const v8h a0 = *(const v8h*)(lrow + o);        // K 0..7 (+8 hi)
        const v8h a1 = *(const v8h*)(lrow + o + 16);   // K 16..23 (+8 hi)
        A[kk] = __builtin_shufflevector(a0, a1, 0, 1, 2, 3, 4, 5, 6, 7,
                                        8, 9, 10, 11, 12, 13, 14, 15);
    }

    const _Float16* rbase = rf16 + (size_t)b * HWH * CC;

    // gather index for offset slot k (this lane's pixel column)
    auto compute_ind = [&](int k) -> int {
        const size_t oidx = (((size_t)b * NUM + k) * HH + py) * WW + px;
        const int dx = (int)offx[oidx];
        const int dy = (int)offy[oidx];
        return clampi(py + dy, 0, HH - 1) * WW + clampi(px + dx, 0, WW - 1);
    };

    // B fragments (32x16 f16): lane (mod 16) = column N = pixel,
    // lanes 0-15 hold K rows {0..7, 16..23}, lanes 16-31 hold K rows {8..15, 24..31}.
    auto load_B = [&](v16h* Bf, int ind) {
        const _Float16* rrow = rbase + (size_t)ind * CC;
#pragma unroll
        for (int kk = 0; kk < 4; ++kk) {
            const int o = 32 * kk + (hi ? 8 : 0);
            const v8h b0 = *(const v8h*)(rrow + o);
            const v8h b1 = *(const v8h*)(rrow + o + 16);
            Bf[kk] = __builtin_shufflevector(b0, b1, 0, 1, 2, 3, 4, 5, 6, 7,
                                             8, 9, 10, 11, 12, 13, 14, 15);
        }
    };

    auto mma_chain = [&](const v16h* Bf) -> v8f {
        v8f acc = {};
#pragma unroll
        for (int kk = 0; kk < 4; ++kk)
            acc = __builtin_amdgcn_wmma_f32_16x16x32_f16(
                false, A[kk], false, Bf[kk], (short)0, acc, false, false);
        return acc;
    };

    // diag(AxB): C/D layout -> lanes 0-7 at element lane, lanes 24-31 at
    // element lane-24 (N = lane mod 16 = pixel within tile).
    auto emit = [&](int k, v8f acc, int ind) {
        inds[((size_t)b * HWH + p) * NUM + k] = ind;   // lanes l, l+16 dup: benign
        const bool valid = (lane < 8) || (lane >= 24);
        const int  ridx  = hi ? (lane - 24) : lane;
        float d = 0.0f;
#pragma unroll
        for (int r = 0; r < 8; ++r)
            if (ridx == r) d = acc[r];
        if (valid) {
            const int n = hi ? (lane - 16) : lane;
            cost[((size_t)b * NUM + k) * HWH + tile * 16 + n] = d;
        }
    };

    v16h B0[4], B1[4];
    int ind0 = compute_ind(0);
    load_B(B0, ind0);

    for (int k = 0; k < NUM; k += 2) {
        const int ind1 = compute_ind(k + 1);           // NUM even: k+1 < NUM
        load_B(B1, ind1);                              // hide behind chain on B0
        emit(k, mma_chain(B0), ind0);

        const int kn = (k + 2 < NUM) ? (k + 2) : (NUM - 1); // tail: redundant, valid
        const int ind2 = compute_ind(kn);
        load_B(B0, ind2);                              // hide behind chain on B1
        emit(k + 1, mma_chain(B1), ind1);
        ind0 = ind2;
    }
}

// ---------------------------------------------------------------------------
// Kernel 2: softmax statistics over the SPATIAL axis (axis=1 of (b,p,k)).
//   One block per (b,k) row of 9216 values: max, then sum(exp(x-max)).
// ---------------------------------------------------------------------------
__global__ __launch_bounds__(256) void k_softmax_stats(
    const float* __restrict__ cost,  // (B, NUM, HW)
    float* __restrict__ stats)       // (B*NUM, 2) = {max, sum}
{
    const int row = blockIdx.x;               // b*NUM + k
    const float* base = cost + (size_t)row * HWH;
    const int tid = threadIdx.x;

    __shared__ float red[256];

    float m = -INFINITY;
    for (int i = tid; i < HWH; i += 256) m = fmaxf(m, base[i]);
    red[tid] = m;
    __syncthreads();
    for (int s = 128; s > 0; s >>= 1) {
        if (tid < s) red[tid] = fmaxf(red[tid], red[tid + s]);
        __syncthreads();
    }
    m = red[0];
    __syncthreads();

    float sum = 0.0f;
    for (int i = tid; i < HWH; i += 256) sum += __expf(base[i] - m);
    red[tid] = sum;
    __syncthreads();
    for (int s = 128; s > 0; s >>= 1) {
        if (tid < s) red[tid] += red[tid + s];
        __syncthreads();
    }
    if (tid == 0) {
        stats[row * 2 + 0] = m;
        stats[row * 2 + 1] = red[0];
    }
}

// ---------------------------------------------------------------------------
// Kernel 3: normalize, per-pixel top-16 (lax.top_k semantics: descending,
//           ties -> lowest index), convert inds back to offsets, emit outputs.
//   out = [ox_out (B,16,HW) | oy_out (B,16,HW) | corr (B,16,HW)]
// ---------------------------------------------------------------------------
__global__ __launch_bounds__(256) void k_topk(
    const float* __restrict__ cost,   // (B, NUM, HW)
    const int*   __restrict__ inds,   // (B, HW, NUM)
    const float* __restrict__ stats,  // (B*NUM, 2)
    float* __restrict__ out)
{
    const int gid = blockIdx.x * 256 + threadIdx.x;
    if (gid >= BB * HWH) return;
    const int b = gid / HWH;
    const int p = gid % HWH;
    const int px = p % WW;
    const int py = p / WW;

    float prob[NUM];
    int   mind[NUM];
#pragma unroll
    for (int k = 0; k < NUM; ++k) {
        const int row = b * NUM + k;
        const float m = stats[row * 2 + 0];
        const float s = stats[row * 2 + 1];
        prob[k] = __expf(cost[(size_t)row * HWH + p] - m) / s;
        mind[k] = inds[((size_t)b * HWH + p) * NUM + k];
    }

    const size_t OFS = (size_t)BB * KOUT * HWH;   // 294912
    unsigned long long used = 0ull;
#pragma unroll
    for (int j = 0; j < KOUT; ++j) {
        float best = -INFINITY;
        int   bi   = 0;
        int   bsel = 0;
#pragma unroll
        for (int k = 0; k < NUM; ++k) {
            const bool free_k = ((used >> k) & 1ull) == 0ull;
            if (free_k && prob[k] > best) {
                best = prob[k];
                bi   = k;
                bsel = mind[k];
            }
        }
        used |= 1ull << bi;

        const float oxv = (float)(bsel % WW - px);
        const float oyv = (float)(bsel / WW - py);
        const size_t o = ((size_t)b * KOUT + j) * HWH + p;
        out[o]            = oxv;
        out[OFS + o]      = oyv;
        out[2 * OFS + o]  = best;
    }
}

// ---------------------------------------------------------------------------
extern "C" void kernel_launch(void* const* d_in, const int* in_sizes, int n_in,
                              void* d_out, int out_size, void* d_ws, size_t ws_size,
                              hipStream_t stream) {
    const float* lf   = (const float*)d_in[0];  // (2,128,9216)
    const float* rf   = (const float*)d_in[1];  // (128, 2*9216)
    const float* offx = (const float*)d_in[2];  // (2,48,96,96)
    const float* offy = (const float*)d_in[3];

    float*    cost  = (float*)d_ws;                              // B*NUM*HW f32
    int*      inds  = (int*)(cost + (size_t)BB * NUM * HWH);     // B*HW*NUM i32
    float*    stats = (float*)(inds + (size_t)BB * HWH * NUM);   // B*NUM*2 f32
    _Float16* lf16  = (_Float16*)(stats + 2 * BB * NUM);         // B*HW*C f16
    _Float16* rf16  = lf16 + (size_t)BB * HWH * CC;              // B*HW*C f16

    k_pack<<<dim3(HWH / TP, BB), 256, 0, stream>>>(lf, rf, lf16, rf16);
    k_cost<<<dim3(HWH / 16, BB), 32, 0, stream>>>(lf16, rf16, offx, offy, cost, inds);
    k_softmax_stats<<<BB * NUM, 256, 0, stream>>>(cost, stats);
    k_topk<<<(BB * HWH + 255) / 256, 256, 0, stream>>>(cost, inds, stats, (float*)d_out);
}
